// Encoder_9242769621272
// MI455X (gfx1250) — matrix operand order repaired and checked by reading, not compiled, not run on hardware
//
#include <hip/hip_runtime.h>

// ---------------------------------------------------------------------------
// Encoder: biLSTM (fwd+bwd) over x[B=64,T=512,N=1024], then LSTM over [T,B,2N].
// bf16 WMMA (v_wmma_f32_16x16x32_bf16) with fp32 accumulation.
// ---------------------------------------------------------------------------

typedef __bf16  bf16_t;
typedef __attribute__((ext_vector_type(16))) __bf16 v16bf;
typedef __attribute__((ext_vector_type(8)))  float  v8f;

#define BSZ   64
#define TSEQ  512
#define NH    1024
#define N4    4096   // 4*NH gate rows

union Frag32B { v16bf v; uint4 q[2]; };

__device__ __forceinline__ v8f wmma_bf16(v16bf a, v16bf b, v8f c) {
  return __builtin_amdgcn_wmma_f32_16x16x32_bf16(false, a, false, b,
                                                 (short)0, c, false, false);
}

// B fragment: 32 contiguous bytes per lane from the packed weight tile.
__device__ __forceinline__ v16bf ld_bfrag(const bf16_t* __restrict__ p) {
  Frag32B f;
  f.q[0] = reinterpret_cast<const uint4*>(p)[0];
  f.q[1] = reinterpret_cast<const uint4*>(p)[1];
  return f.v;
}

// A fragment: per ISA 7.12.2 (16-bit A 16x32) lane holds K = {base..base+7,
// base+16..base+23}: two 16B loads with a 16B gap.
__device__ __forceinline__ v16bf ld_afrag(const bf16_t* __restrict__ p) {
  Frag32B f;
  f.q[0] = *reinterpret_cast<const uint4*>(p);
  f.q[1] = *reinterpret_cast<const uint4*>(p + 16);
  return f.v;
}

// ---------------------------------------------------------------------------
// Branch-free GEMM segment: nkt K-tiles (nkt even), two M-tiles per wave
// sharing each B fragment. Two K-tiles per iteration + unroll so next-tile
// loads are in flight while current WMMAs execute.
// ---------------------------------------------------------------------------
__device__ __forceinline__ void gemm_seg(
    const bf16_t* __restrict__ pB,     // packed tiles, already at lane offset
    const bf16_t* __restrict__ a0p,    // A row  (m2*32 + mrow), at lane K offset
    const bf16_t* __restrict__ a1p,    // A row +16
    int nkt, v8f& acc0, v8f& acc1) {
#pragma unroll 2
  for (int kt = 0; kt < nkt; kt += 2) {
    const bf16_t* B0 = pB + (size_t)kt * 512;
    v16bf b0 = ld_bfrag(B0);
    v16bf b1 = ld_bfrag(B0 + 512);
    v16bf p0 = ld_afrag(a0p + kt * 32);
    v16bf q0 = ld_afrag(a1p + kt * 32);
    v16bf p1 = ld_afrag(a0p + kt * 32 + 32);
    v16bf q1 = ld_afrag(a1p + kt * 32 + 32);
    __builtin_prefetch(B0 + 512 * 12, 0, 3);   // pull upcoming weight tiles to WGP$
    acc0 = wmma_bf16(p0, b0, acc0);
    acc1 = wmma_bf16(q0, b0, acc1);
    acc0 = wmma_bf16(p1, b1, acc0);
    acc1 = wmma_bf16(q1, b1, acc1);
  }
}

// ---------------------------------------------------------------------------
// gates = [A0 | A1] x W^T for one 16-wide hidden slice, all 4 gates.
// 8 waves: wave = (gate[0..3], mhalf[0..1]); each wave owns a 32(M)x16(N)
// region = 2 accumulators. A1 == nullptr means h_{-1} == 0: segment skipped.
// Result spilled to LDS gbuf[4][64][16] per the C-matrix VGPR layout.
// ---------------------------------------------------------------------------
__device__ __forceinline__ void lstm_gates_gemm(
    const bf16_t* __restrict__ A0, int ld0, int KT0,
    const bf16_t* __restrict__ A1, int ld1, int KT1,
    const bf16_t* __restrict__ P,  int j0,  int tid,
    float* __restrict__ gbuf) {
  const int lane  = tid & 31;
  const int wave  = tid >> 5;
  const int gate  = wave >> 1;
  const int m2    = wave & 1;
  const int lhalf = lane >> 4;
  const int mrow  = lane & 15;
  const int col   = lane & 15;
  const int KT    = KT0 + KT1;
  const int ntile = gate * (NH >> 4) + (j0 >> 4);
  const bf16_t* pB = P + ((size_t)ntile * KT) * 512 + lane * 16;

  v8f acc0 = {};
  v8f acc1 = {};

  {
    const bf16_t* a0p = A0 + (size_t)(m2 * 32 + mrow) * ld0 + lhalf * 8;
    gemm_seg(pB, a0p, a0p + 16 * ld0, KT0, acc0, acc1);
  }
  if (A1) {
    const bf16_t* a0p = A1 + (size_t)(m2 * 32 + mrow) * ld1 + lhalf * 8;
    gemm_seg(pB + (size_t)KT0 * 512, a0p, a0p + 16 * ld1, KT1, acc0, acc1);
  }

  // C layout (32-bit C/D 16x16): lane l, vgpr v -> (m = 8*(l>>4)+v, n = l&15)
  float* g = gbuf + (size_t)gate * 64 * 16;
#pragma unroll
  for (int v = 0; v < 8; ++v) {
    g[(m2 * 32 +      lhalf * 8 + v) * 16 + col] = acc0[v];
    g[(m2 * 32 + 16 + lhalf * 8 + v) * 16 + col] = acc1[v];
  }
}

__device__ __forceinline__ float sigm(float x) { return 1.0f / (1.0f + expf(-x)); }

// ---------------------------------------------------------------------------
// BiLSTM step t: blocks 0..63 = forward @ time t, blocks 64..127 = backward
// @ time 511-t. Each block owns hidden slice [j0, j0+16).
// ---------------------------------------------------------------------------
__global__ void __launch_bounds__(256)
k_bilstm_step(const bf16_t* __restrict__ Xbf, bf16_t* __restrict__ bi,
              const bf16_t* __restrict__ Pfwd, const bf16_t* __restrict__ Pbwd,
              const float* __restrict__ bias_f, const float* __restrict__ bias_b,
              float* __restrict__ cf, float* __restrict__ cb, int t) {
  __shared__ float gbuf[4 * 64 * 16];
  const int dir   = blockIdx.x >> 6;
  const int slice = blockIdx.x & 63;
  const int j0    = slice << 4;
  const int tt    = dir ? (TSEQ - 1 - t) : t;

  const bf16_t* A0 = Xbf + (size_t)tt * BSZ * NH;
  const bf16_t* A1 = nullptr;
  if (t > 0) {
    const int tprev = dir ? (tt + 1) : (tt - 1);
    A1 = bi + (size_t)tprev * BSZ * (2 * NH) + (dir ? NH : 0);
  }
  const bf16_t* P    = dir ? Pbwd   : Pfwd;
  const float*  bias = dir ? bias_b : bias_f;
  float*        cbuf = dir ? cb     : cf;

  lstm_gates_gemm(A0, NH, NH >> 5, A1, 2 * NH, NH >> 5, P, j0, threadIdx.x, gbuf);
  __syncthreads();

#pragma unroll
  for (int it = 0; it < 4; ++it) {
    const int idx = threadIdx.x + (it << 8);   // 0..1023 = (b, jj)
    const int b = idx >> 4, jj = idx & 15;
    const int j = j0 + jj;
    float gi = gbuf[(0 * 64 + b) * 16 + jj] + bias[j];
    float gf = gbuf[(1 * 64 + b) * 16 + jj] + bias[NH + j];
    float gg = gbuf[(2 * 64 + b) * 16 + jj] + bias[2 * NH + j];
    float go = gbuf[(3 * 64 + b) * 16 + jj] + bias[3 * NH + j];
    gi = sigm(gi); gf = sigm(gf); gg = tanhf(gg); go = sigm(go);
    const float c  = (t == 0) ? 0.0f : cbuf[b * NH + j];
    const float cn = gf * c + gi * gg;
    cbuf[b * NH + j] = cn;
    const float hn = go * tanhf(cn);
    bi[(size_t)tt * BSZ * (2 * NH) + (size_t)b * (2 * NH) + dir * NH + j] = (bf16_t)hn;
  }
}

// ---------------------------------------------------------------------------
// Cell LSTM step t over bi_out (K = 2N input + N recurrent = 3072).
// 64 blocks, one per 16-wide hidden slice; h ping-pongs across launches.
// ---------------------------------------------------------------------------
__global__ void __launch_bounds__(256)
k_cell_step(const bf16_t* __restrict__ bi, const bf16_t* __restrict__ Pcell,
            const float* __restrict__ bias_c, float* __restrict__ cc,
            const bf16_t* __restrict__ hprev, bf16_t* __restrict__ hnext,
            float* __restrict__ out, int t) {
  __shared__ float gbuf[4 * 64 * 16];
  const int j0 = (int)blockIdx.x << 4;
  const bf16_t* A0 = bi + (size_t)t * BSZ * (2 * NH);

  lstm_gates_gemm(A0, 2 * NH, (2 * NH) >> 5, hprev, NH, NH >> 5, Pcell, j0,
                  threadIdx.x, gbuf);
  __syncthreads();

#pragma unroll
  for (int it = 0; it < 4; ++it) {
    const int idx = threadIdx.x + (it << 8);
    const int b = idx >> 4, jj = idx & 15;
    const int j = j0 + jj;
    float gi = gbuf[(0 * 64 + b) * 16 + jj] + bias_c[j];
    float gf = gbuf[(1 * 64 + b) * 16 + jj] + bias_c[NH + j];
    float gg = gbuf[(2 * 64 + b) * 16 + jj] + bias_c[2 * NH + j];
    float go = gbuf[(3 * 64 + b) * 16 + jj] + bias_c[3 * NH + j];
    gi = sigm(gi); gf = sigm(gf); gg = tanhf(gg); go = sigm(go);
    const float c  = (t == 0) ? 0.0f : cc[b * NH + j];
    const float cn = gf * c + gi * gg;
    cc[b * NH + j] = cn;
    const float hn = go * tanhf(cn);
    hnext[b * NH + j] = (bf16_t)hn;
    out[(size_t)b * TSEQ * NH + (size_t)t * NH + j] = hn;
    if (t == TSEQ - 1) {
      out[(size_t)BSZ * TSEQ * NH + (size_t)b * NH + j] = hn;                    // hT
      out[(size_t)BSZ * TSEQ * NH + (size_t)BSZ * NH + (size_t)b * NH + j] = cn; // cT
    }
  }
}

// ---------------------------------------------------------------------------
// Prep: cast x [B,T,N] fp32 -> Xbf [T,B,N] bf16 (time-major for step GEMMs).
// ---------------------------------------------------------------------------
__global__ void __launch_bounds__(256)
k_cast_x(const float* __restrict__ src, bf16_t* __restrict__ dst) {
  const size_t i = (size_t)blockIdx.x * blockDim.x + threadIdx.x;
  const int n = (int)(i & (NH - 1));
  const int b = (int)((i >> 10) & (BSZ - 1));
  const int t = (int)(i >> 16);                // NH*BSZ = 2^16
  dst[i] = (bf16_t)src[((size_t)b * TSEQ + t) * NH + n];
}

// ---------------------------------------------------------------------------
// Prep: pack fused [W_ih ; W_hh] into 32(K)x16(N) B-fragment tiles, lane-
// contiguous: lane = ((klocal/8)&1)*16 + nlocal, idx = (klocal&7) + (klocal/16)*8.
// Tiles laid out [ntile][ktile] so the K loop streams contiguously.
// ---------------------------------------------------------------------------
__global__ void __launch_bounds__(256)
k_pack_w(const float* __restrict__ Wih, int K1,
         const float* __restrict__ Whh, int K2, bf16_t* __restrict__ dst) {
  const int Ktot = K1 + K2;
  const size_t e = (size_t)blockIdx.x * blockDim.x + threadIdx.x;
  if (e >= (size_t)N4 * Ktot) return;
  const int n = (int)(e / Ktot);
  const int k = (int)(e % Ktot);
  const float val = (k < K1) ? Wih[(size_t)n * K1 + k]
                             : Whh[(size_t)n * K2 + (k - K1)];
  const int KT = Ktot >> 5;
  const int ntile = n >> 4, nl = n & 15;
  const int kt = k >> 5,  kl = k & 31;
  const int pos = ((((kl >> 3) & 1) << 4) + nl) * 16 + ((kl & 7) + ((kl >> 4) << 3));
  dst[(size_t)(ntile * KT + kt) * 512 + pos] = (bf16_t)val;
}

// ---------------------------------------------------------------------------
extern "C" void kernel_launch(void* const* d_in, const int* in_sizes, int n_in,
                              void* d_out, int out_size, void* d_ws, size_t ws_size,
                              hipStream_t stream) {
  (void)in_sizes; (void)n_in; (void)out_size; (void)ws_size;
  const float* x     = (const float*)d_in[0];
  const float* Wf_ih = (const float*)d_in[1];
  const float* Wf_hh = (const float*)d_in[2];
  const float* bfv   = (const float*)d_in[3];
  const float* Wb_ih = (const float*)d_in[4];
  const float* Wb_hh = (const float*)d_in[5];
  const float* bbv   = (const float*)d_in[6];
  const float* Wc_ih = (const float*)d_in[7];
  const float* Wc_hh = (const float*)d_in[8];
  const float* bcv   = (const float*)d_in[9];
  float* out = (float*)d_out;

  // Workspace carve-up (~250 MB). Packed buffers padded 16 KB for the
  // unguarded prefetch run-ahead.
  char* ws = (char*)d_ws;
  size_t o = 0;
  auto carve = [&](size_t bytes) {
    void* p = ws + o;
    o += (bytes + 255) & ~(size_t)255;
    return p;
  };
  const size_t PAD = 16384;
  bf16_t* Xbf   = (bf16_t*)carve((size_t)TSEQ * BSZ * NH * 2);              //  64 MB
  bf16_t* bi    = (bf16_t*)carve((size_t)TSEQ * BSZ * 2 * NH * 2);          // 128 MB
  bf16_t* Pfwd  = (bf16_t*)carve((size_t)N4 * 2 * NH * 2 + PAD);            //  16 MB
  bf16_t* Pbwd  = (bf16_t*)carve((size_t)N4 * 2 * NH * 2 + PAD);            //  16 MB
  bf16_t* Pcell = (bf16_t*)carve((size_t)N4 * 3 * NH * 2 + PAD);            //  24 MB
  float*  cf    = (float*)carve((size_t)BSZ * NH * 4);
  float*  cb    = (float*)carve((size_t)BSZ * NH * 4);
  float*  cc    = (float*)carve((size_t)BSZ * NH * 4);
  bf16_t* hc0   = (bf16_t*)carve((size_t)BSZ * NH * 2);
  bf16_t* hc1   = (bf16_t*)carve((size_t)BSZ * NH * 2);
  bf16_t* hbufs[2] = {hc0, hc1};

  // One-time prep.
  k_cast_x<<<dim3((TSEQ * BSZ * NH) / 256), dim3(256), 0, stream>>>(x, Xbf);
  k_pack_w<<<dim3(((size_t)N4 * 2048 + 255) / 256), dim3(256), 0, stream>>>(
      Wf_ih, NH, Wf_hh, NH, Pfwd);
  k_pack_w<<<dim3(((size_t)N4 * 2048 + 255) / 256), dim3(256), 0, stream>>>(
      Wb_ih, NH, Wb_hh, NH, Pbwd);
  k_pack_w<<<dim3(((size_t)N4 * 3072 + 255) / 256), dim3(256), 0, stream>>>(
      Wc_ih, 2 * NH, Wc_hh, NH, Pcell);

  // Serial recurrences: kernel boundary = global sync between timesteps.
  for (int t = 0; t < TSEQ; ++t)
    k_bilstm_step<<<dim3(128), dim3(256), 0, stream>>>(
        Xbf, bi, Pfwd, Pbwd, bfv, bbv, cf, cb, t);

  for (int t = 0; t < TSEQ; ++t) {
    const bf16_t* hprev = (t == 0) ? nullptr : hbufs[(t + 1) & 1];
    bf16_t* hnext = hbufs[t & 1];
    k_cell_step<<<dim3(64), dim3(256), 0, stream>>>(
        bi, Pcell, bcv, cc, hprev, hnext, out, t);
  }
}
